// IronwoodTernaryLayer_71287867179134
// MI455X (gfx1250) — matrix operand order repaired and checked by reading, not compiled, not run on hardware
//
#include <hip/hip_runtime.h>
#include <hip/hip_bf16.h>

// ---------- vector typedefs matching the gfx1250 WMMA builtin signatures ----------
typedef __bf16 v16bf __attribute__((ext_vector_type(16)));
typedef __bf16 v8bf  __attribute__((ext_vector_type(8)));
typedef __bf16 v4bf  __attribute__((ext_vector_type(4)));
typedef float  v8f   __attribute__((ext_vector_type(8)));
typedef float  v4f   __attribute__((ext_vector_type(4)));

#define M_DIM 8192
#define N_DIM 4096
#define K_DIM 4096
#define NELEM_W ((size_t)N_DIM * (size_t)K_DIM)   // 16,777,216

#define BM 128
#define BN 128
#define BK 32
#define LDP 40            // padded LDS row stride in bf16 elems (80B, 16B aligned, spreads banks)
#define BUFSZ (BM * LDP)  // one tile buffer, elements

// ---------------------------------------------------------------------
// Packed fp32 -> 2x bf16 conversion (RTE) via the native CDNA5 VALU op.
// The compiler only emits the scalar form (v_cvt_pk_bf16_f32 d, s, 0) +
// v_mov_b16 repack chains; force the true packing form.
// ---------------------------------------------------------------------
__device__ __forceinline__ unsigned cvt_pk_bf16(float a, float b) {
  unsigned r;
  asm("v_cvt_pk_bf16_f32 %0, %1, %2" : "=v"(r) : "v"(a), "v"(b));
  return r;   // bf16(a) in [15:0], bf16(b) in [31:16]
}

// split two fp32 into packed bf16 hi + packed bf16 residual (x = hi + lo)
__device__ __forceinline__ void split_pair(float a, float b,
                                           unsigned& hp, unsigned& lp) {
  hp = cvt_pk_bf16(a, b);
  const float h0 = __uint_as_float(hp << 16);          // fp32 image of bf16(a)
  const float h1 = __uint_as_float(hp & 0xFFFF0000u);  // fp32 image of bf16(b)
  lp = cvt_pk_bf16(a - h0, b - h1);
}

// ======================================================================
// Kernel 1: block-partial sum of |W|  (fixed-tree, deterministic)
// ======================================================================
__global__ __launch_bounds__(256) void absum_partial_kernel(
    const float* __restrict__ w, float* __restrict__ partials) {
  __shared__ float red[256];
  const int tid = threadIdx.x;
  const size_t n4 = NELEM_W / 4;
  const size_t stride = (size_t)gridDim.x * 256;
  const v4f* w4 = (const v4f*)w;
  float s = 0.0f;
  for (size_t i = (size_t)blockIdx.x * 256 + tid; i < n4; i += stride) {
    v4f v = w4[i];
    s += fabsf(v[0]) + fabsf(v[1]) + fabsf(v[2]) + fabsf(v[3]);
  }
  red[tid] = s;
  __syncthreads();
  for (int off = 128; off > 0; off >>= 1) {
    if (tid < off) red[tid] += red[tid + off];
    __syncthreads();
  }
  if (tid == 0) partials[blockIdx.x] = red[0];
}

// ======================================================================
// Kernel 2: finalize gamma = sum/n ; zero the 3 bin counters
// ======================================================================
__global__ __launch_bounds__(256) void finalize_gamma_kernel(
    const float* __restrict__ partials, int n_part,
    float* __restrict__ gamma_out, int* __restrict__ counts) {
  __shared__ float red[256];
  const int tid = threadIdx.x;
  float s = 0.0f;
  for (int i = tid; i < n_part; i += 256) s += partials[i];
  red[tid] = s;
  __syncthreads();
  for (int off = 128; off > 0; off >>= 1) {
    if (tid < off) red[tid] += red[tid + off];
    __syncthreads();
  }
  if (tid == 0) gamma_out[0] = red[0] / (float)NELEM_W;
  if (tid < 3) counts[tid] = 0;
}

// ======================================================================
// Kernel 3: ternary quantize W -> bf16 (exact for {-1,0,1}) + bin counts
// round-half-to-even via rintf (default RTE) matches jnp.round
// ======================================================================
__global__ __launch_bounds__(256) void quantize_kernel(
    const float* __restrict__ w, const float* __restrict__ gamma_p,
    __bf16* __restrict__ wq, int* __restrict__ counts) {
  __shared__ int c[3];
  const int tid = threadIdx.x;
  if (tid < 3) c[tid] = 0;
  __syncthreads();

  const float inv = 1.0f / (gamma_p[0] + 1e-8f);
  const size_t i4 = (size_t)blockIdx.x * 256 + tid;
  v4f v = ((const v4f*)w)[i4];
  float r[4];
  int cm = 0, cz = 0, cp = 0;
#pragma unroll
  for (int j = 0; j < 4; ++j) {
    float t = rintf(v[j] * inv);
    t = fminf(1.0f, fmaxf(-1.0f, t));
    r[j] = t;
    cm += (t == -1.0f);
    cz += (t == 0.0f);
    cp += (t == 1.0f);
  }
  uint2 q;
  q.x = cvt_pk_bf16(r[0], r[1]);   // exact for {-1,0,1}
  q.y = cvt_pk_bf16(r[2], r[3]);
  ((uint2*)wq)[i4] = q;

  atomicAdd(&c[0], cm);
  atomicAdd(&c[1], cz);
  atomicAdd(&c[2], cp);
  __syncthreads();
  if (tid < 3) atomicAdd(&counts[tid], c[tid]);
}

// ======================================================================
// Kernel 4: entropy + thermal scalars -> tail of d_out
// ======================================================================
__global__ void finalize_scalars_kernel(const int* __restrict__ counts,
                                        float* __restrict__ out_tail) {
  if (threadIdx.x == 0 && blockIdx.x == 0) {
    const float n = (float)NELEM_W;
    float H = 0.0f;
#pragma unroll
    for (int i = 0; i < 3; ++i) {
      float p = (float)counts[i] / n;
      if (p > 0.0f) H -= p * log2f(p);
    }
    const float heat = 100.0f * fmaxf(H - 0.0f, 0.0f);   // ALPHA * relu(H - H_BASELINE)
    const float q_total = 50.0f + heat;                  // P_BASE + heat
    out_tail[0] = -10.0f + q_total * 0.5f;               // T_AMBIENT + q*R_TBR
    out_tail[1] = H;
  }
}

// ======================================================================
// Kernel 5: out[M,N] = x[M,K] @ Wq[N,K]^T with split-bf16 WMMA
//   128x128x32 block tile, 8 waves; double-buffered LDS, software-pipelined
//   staging (one barrier per K-step), packed v_cvt_pk_bf16_f32 splits,
//   non-temporal output stores so the L2-resident x / Wq working set
//   (~168MB < 192MB L2) is not evicted by the write-once output stream.
// ======================================================================
__global__ __launch_bounds__(256) void ternary_gemm_kernel(
    const float* __restrict__ x, const __bf16* __restrict__ wq,
    float* __restrict__ out) {
  __shared__ __align__(16) __bf16 sXhi[2 * BUFSZ];
  __shared__ __align__(16) __bf16 sXlo[2 * BUFSZ];
  __shared__ __align__(16) __bf16 sW [2 * BUFSZ];

  const int tid   = threadIdx.x;
  const int wave  = tid >> 5;
  const int lane  = tid & 31;
  const int ln16  = lane & 15;
  const int khalf = lane >> 4;

  const int n0 = blockIdx.x * BN;
  const int m0 = blockIdx.y * BM;

  v8f acc[8] = {};

  // ---- per-thread staging addresses (fixed row/col; advance by BK per iter) ----
  // x tile: 128 rows x 32 cols fp32 = 1024 float4 -> 4 float4 per thread
  const float* xg[4];
  int xoff[4];
#pragma unroll
  for (int i = 0; i < 4; ++i) {
    const int idx = i * 256 + tid;
    const int row = idx >> 3;        // 8 float4 per 32-float row
    const int c4  = idx & 7;
    xg[i]   = x + (size_t)(m0 + row) * K_DIM + (size_t)c4 * 4;
    xoff[i] = row * LDP + c4 * 4;
  }
  // w tile: 128 rows x 32 cols bf16 = 512 v8bf -> 2 v8bf per thread
  const __bf16* wg[2];
  int woff[2];
#pragma unroll
  for (int i = 0; i < 2; ++i) {
    const int idx = i * 256 + tid;
    const int row = idx >> 2;        // 4 v8bf per 32-bf16 row
    const int c8  = idx & 3;
    wg[i]   = wq + (size_t)(n0 + row) * K_DIM + (size_t)c8 * 8;
    woff[i] = row * LDP + c8 * 8;
  }

  // compute-side LDS offsets
  const int aOff = (wave * 16 + ln16) * LDP;           // A row base
  // pipeline registers for the next tile
  v4f  xr[4];
  v8bf wr[2];

  // ---- prologue: fetch tile 0, convert (packed), store into buffer 0 ----
#pragma unroll
  for (int i = 0; i < 4; ++i) xr[i] = *(const v4f*)(xg[i]);
#pragma unroll
  for (int i = 0; i < 2; ++i) wr[i] = *(const v8bf*)(wg[i]);
#pragma unroll
  for (int i = 0; i < 4; ++i) {
    uint2 hv, lv;
    split_pair(xr[i][0], xr[i][1], hv.x, lv.x);
    split_pair(xr[i][2], xr[i][3], hv.y, lv.y);
    *(uint2*)&sXhi[xoff[i]] = hv;
    *(uint2*)&sXlo[xoff[i]] = lv;
  }
#pragma unroll
  for (int i = 0; i < 2; ++i) *(v8bf*)&sW[woff[i]] = wr[i];
  __syncthreads();

  const int KTILES = K_DIM / BK;
  for (int kk = 0; kk < KTILES; ++kk) {
    const int p    = kk & 1;
    const int base = p * BUFSZ;
    const bool more = (kk + 1 < KTILES);

    // ---- issue next tile's global loads first (overlap with WMMAs) ----
    if (more) {
      const int k1 = (kk + 1) * BK;
#pragma unroll
      for (int i = 0; i < 4; ++i) {
        xr[i] = *(const v4f*)(xg[i] + k1);
        __builtin_prefetch((const void*)(xg[i] + k1 + BK), 0, 1);
      }
#pragma unroll
      for (int i = 0; i < 2; ++i) {
        wr[i] = *(const v8bf*)(wg[i] + k1);
        __builtin_prefetch((const void*)(wg[i] + k1 + BK), 0, 1);
      }
    }

    // ---- A operand (ISA 16-bit A 16x32 layout): lane holds K = 8*khalf..+7 and 16+8*khalf..+7
    const __bf16* aHiBase = &sXhi[base + aOff];
    const __bf16* aLoBase = &sXlo[base + aOff];
    v8bf ah0 = *(const v8bf*)(aHiBase + 8 * khalf);
    v8bf ah1 = *(const v8bf*)(aHiBase + 16 + 8 * khalf);
    v16bf a_hi = __builtin_shufflevector(ah0, ah1, 0, 1, 2, 3, 4, 5, 6, 7,
                                         8, 9, 10, 11, 12, 13, 14, 15);
    v8bf al0 = *(const v8bf*)(aLoBase + 8 * khalf);
    v8bf al1 = *(const v8bf*)(aLoBase + 16 + 8 * khalf);
    v16bf a_lo = __builtin_shufflevector(al0, al1, 0, 1, 2, 3, 4, 5, 6, 7,
                                         8, 9, 10, 11, 12, 13, 14, 15);

    // ---- B operand (32x16): lane (n,khalf) holds K = 8*khalf..+7 and 16+8*khalf..+7 for column n
#pragma unroll
    for (int j = 0; j < 8; ++j) {
      const __bf16* bBase = &sW[base + (j * 16 + ln16) * LDP + 8 * khalf];
      v8bf b0 = *(const v8bf*)(bBase);
      v8bf b1 = *(const v8bf*)(bBase + 16);
      v16bf b = __builtin_shufflevector(b0, b1, 0, 1, 2, 3, 4, 5, 6, 7,
                                        8, 9, 10, 11, 12, 13, 14, 15);
      acc[j] = __builtin_amdgcn_wmma_f32_16x16x32_bf16(
          false, a_hi, false, b, (short)0, acc[j], false, false);
      acc[j] = __builtin_amdgcn_wmma_f32_16x16x32_bf16(
          false, a_lo, false, b, (short)0, acc[j], false, false);
    }

    // ---- convert (packed) + store next tile into the other buffer ----
    if (more) {
      const int nbase = (p ^ 1) * BUFSZ;
#pragma unroll
      for (int i = 0; i < 4; ++i) {
        uint2 hv, lv;
        split_pair(xr[i][0], xr[i][1], hv.x, lv.x);
        split_pair(xr[i][2], xr[i][3], hv.y, lv.y);
        *(uint2*)&sXhi[nbase + xoff[i]] = hv;
        *(uint2*)&sXlo[nbase + xoff[i]] = lv;
      }
#pragma unroll
      for (int i = 0; i < 2; ++i) *(v8bf*)&sW[nbase + woff[i]] = wr[i];
    }
    __syncthreads();   // next buffer ready AND this buffer fully consumed
  }

  // ---- store: D vgpr v, lanes0-15 -> M=v, lanes16-31 -> M=v+8 ; N = lane&15
  // non-temporal: output is write-once; keep x/Wq resident in the 192MB L2.
  float* outBase = out + (size_t)(m0 + wave * 16 + khalf * 8) * N_DIM + n0 + ln16;
#pragma unroll
  for (int j = 0; j < 8; ++j) {
#pragma unroll
    for (int v = 0; v < 8; ++v) {
      __builtin_nontemporal_store(acc[j][v], &outBase[(size_t)v * N_DIM + j * 16]);
    }
  }
}

// ======================================================================
// launcher
// ======================================================================
extern "C" void kernel_launch(void* const* d_in, const int* in_sizes, int n_in,
                              void* d_out, int out_size, void* d_ws, size_t ws_size,
                              hipStream_t stream) {
  (void)in_sizes; (void)n_in; (void)out_size; (void)ws_size;

  const float* x = (const float*)d_in[0];   // [8192, 4096] fp32
  const float* w = (const float*)d_in[1];   // [4096, 4096] fp32
  float* out = (float*)d_out;               // [8192*4096] + t + H

  uint8_t* ws = (uint8_t*)d_ws;
  float* gamma    = (float*)ws;             // 4 B
  int*   counts   = (int*)(ws + 16);        // 12 B
  float* partials = (float*)(ws + 64);      // 2048 * 4 B
  __bf16* wq      = (__bf16*)(ws + 16384);  // 4096*4096*2 B = 32 MB

  const int NPART = 2048;
  absum_partial_kernel<<<NPART, 256, 0, stream>>>(w, partials);
  finalize_gamma_kernel<<<1, 256, 0, stream>>>(partials, NPART, gamma, counts);
  quantize_kernel<<<(int)(NELEM_W / 4 / 256), 256, 0, stream>>>(w, gamma, wq, counts);
  finalize_scalars_kernel<<<1, 64, 0, stream>>>(counts, out + NELEM_W);

  dim3 grid(N_DIM / BN, M_DIM / BM);   // 32 x 64 blocks
  ternary_gemm_kernel<<<grid, 256, 0, stream>>>(x, wq, out);
}
